// SelfAttention_21990232555727
// MI455X (gfx1250) — matrix operand order repaired and checked by reading, not compiled, not run on hardware
//
#include <hip/hip_runtime.h>

// ---------------------------------------------------------------------------
// Fused causal self-attention (QKV proj + RoPE + flash attention + out proj)
// for MI455X / gfx1250: bf16 WMMA (v_wmma_f32_16x16x32_bf16), f32 accumulate.
// B=2, T=2048, C=2048, H=16, HD=128.
// ---------------------------------------------------------------------------

typedef __attribute__((ext_vector_type(16))) __bf16 v16bf;
typedef __attribute__((ext_vector_type(8)))  __bf16 v8bf;
typedef __attribute__((ext_vector_type(8)))  float  v8f;

#define DEV __device__ __forceinline__

constexpr int B_    = 2;
constexpr int T_    = 2048;
constexpr int C_    = 2048;
constexpr int H_    = 16;
constexpr int HD_   = 128;
constexpr int BT_   = B_ * T_;     // 4096 rows
constexpr int NQKV_ = 3 * C_;      // 6144

DEV unsigned short f2bf(float f) {            // round-to-nearest-even f32->bf16
  unsigned int u = __float_as_uint(f);
  u += 0x7FFFu + ((u >> 16) & 1u);
  return (unsigned short)(u >> 16);
}

DEV v8f vzero8() {
  v8f v;
#pragma unroll
  for (int r = 0; r < 8; ++r) v[r] = 0.0f;
  return v;
}

// A fragment (16x32 bf16, MxK): lane<16 holds row M=lane, K 0..15;
// lane>=16 holds row M=lane-16, K 16..31. p = &A[row*ldk + kbase].
DEV v16bf load_a16(const unsigned short* p) {
  union { v16bf v; v8bf h[2]; } u;
  u.h[0] = *(const v8bf*)(p);
  u.h[1] = *(const v8bf*)(p + 8);
  return u.v;
}
// B fragment (32x16 bf16, KxN stored as [N][K] row-major): lane = column,
// 16 contiguous K values; lane<16 -> K 0..15, lane>=16 -> K 16..31.
DEV v16bf load_b16(const unsigned short* p) { return *(const v16bf*)(p); }

DEV v8f wmma_bf16(v16bf a, v16bf b, v8f c) {
  return __builtin_amdgcn_wmma_f32_16x16x32_bf16(
      /*neg_a=*/false, a, /*neg_b=*/false, b,
      /*c_mod=*/(short)0, c, /*reuse_a=*/false, /*reuse_b=*/false);
}

// ---------------------------------------------------------------------------
// f32 -> bf16 elementwise convert
// ---------------------------------------------------------------------------
__global__ __launch_bounds__(256)
void cvt_f32_bf16(const float* __restrict__ in, unsigned short* __restrict__ out, int n) {
  int i = blockIdx.x * blockDim.x + threadIdx.x;
  if (i < n) out[i] = f2bf(in[i]);
}

// ---------------------------------------------------------------------------
// GEMM: D[M,N] (f32) = A[M,K] (bf16, row-major) * B[N,K]^T (bf16, row-major).
// 256 threads = 8 waves in 2x4; block tile 128x256; wave tile 64x64
// (16 f32 16x16 accumulators); K-step 32 -> 16 WMMAs per step.
// ---------------------------------------------------------------------------
__global__ __launch_bounds__(256)
void gemm_nt_bf16(const unsigned short* __restrict__ A,
                  const unsigned short* __restrict__ Bm,
                  float* __restrict__ D, int M, int N, int K) {
  const int lane = threadIdx.x & 31;
  const int wave = threadIdx.x >> 5;
  const int half = lane >> 4, ln = lane & 15;
  const int m0 = blockIdx.y * 128 + (wave >> 2) * 64;
  const int n0 = blockIdx.x * 256 + (wave & 3) * 64;

  v8f acc[4][4];
#pragma unroll
  for (int i = 0; i < 4; ++i)
#pragma unroll
    for (int j = 0; j < 4; ++j) acc[i][j] = vzero8();

  const unsigned short* aP[4];
  const unsigned short* bP[4];
#pragma unroll
  for (int i = 0; i < 4; ++i)
    aP[i] = A + (size_t)(m0 + i * 16 + ln) * K + half * 16;
#pragma unroll
  for (int j = 0; j < 4; ++j)
    bP[j] = Bm + (size_t)(n0 + j * 16 + ln) * K + half * 16;

  for (int k0 = 0; k0 < K; k0 += 32) {
    v16bf af[4], bfrag[4];
#pragma unroll
    for (int i = 0; i < 4; ++i) af[i] = load_a16(aP[i] + k0);
#pragma unroll
    for (int j = 0; j < 4; ++j) bfrag[j] = load_b16(bP[j] + k0);
#pragma unroll
    for (int i = 0; i < 4; ++i)
#pragma unroll
      for (int j = 0; j < 4; ++j)
        acc[i][j] = wmma_bf16(af[i], bfrag[j], acc[i][j]);
  }

  // D-tile layout: VGPR r, lanes 0-15 -> row r, lanes 16-31 -> row r+8.
#pragma unroll
  for (int i = 0; i < 4; ++i) {
    const int row0 = m0 + i * 16 + half * 8;
#pragma unroll
    for (int j = 0; j < 4; ++j) {
      const int col = n0 + j * 16 + ln;
#pragma unroll
      for (int r = 0; r < 8; ++r)
        D[(size_t)(row0 + r) * N + col] = acc[i][j][r];
    }
  }
}

// ---------------------------------------------------------------------------
// RoPE + repack: qkv f32 [B*T, 3C] -> Q,K bf16 [b,h,t,d] and V^T bf16 [b,h,d,t]
// One thread per (b,t,h,i) rotation pair; i in [0, HD/2).
// ---------------------------------------------------------------------------
__global__ __launch_bounds__(256)
void rope_pack(const float* __restrict__ qkv,
               unsigned short* __restrict__ Qb,
               unsigned short* __restrict__ Kb,
               unsigned short* __restrict__ Vt) {
  int idx = blockIdx.x * blockDim.x + threadIdx.x;
  if (idx >= B_ * T_ * H_ * (HD_ / 2)) return;
  const int i = idx & 63;            // HD/2 = 64
  const int h = (idx >> 6) & 15;
  const int t = (idx >> 10) & 2047;
  const int b = idx >> 21;

  // theta_i = 10000^(-2i/HD); ln(10000) = 9.210340371976184
  const float freq = __expf(-(float)(2 * i) * (9.210340371976184f / (float)HD_));
  const float ang = (float)t * freq;
  const float cs = __cosf(ang), sn = __sinf(ang);

  const size_t m = (size_t)(b * T_ + t) * NQKV_;
  const int coff = h * HD_ + 2 * i;
  const float q0 = qkv[m + coff],          q1 = qkv[m + coff + 1];
  const float k0 = qkv[m + C_ + coff],     k1 = qkv[m + C_ + coff + 1];
  const float v0 = qkv[m + 2 * C_ + coff], v1 = qkv[m + 2 * C_ + coff + 1];

  const int bh = b * H_ + h;
  const size_t qo = (size_t)(bh * T_ + t) * HD_ + 2 * i;
  Qb[qo]     = f2bf(q0 * cs - q1 * sn);
  Qb[qo + 1] = f2bf(q1 * cs + q0 * sn);
  Kb[qo]     = f2bf(k0 * cs - k1 * sn);
  Kb[qo + 1] = f2bf(k1 * cs + k0 * sn);
  const size_t vo = ((size_t)bh * HD_ + 2 * i) * T_ + t;  // transposed
  Vt[vo]      = f2bf(v0);
  Vt[vo + T_] = f2bf(v1);
}

// ---------------------------------------------------------------------------
// Flash attention: one wave per 16-query block of one (b,h). Key tiles of 32.
// S = Q K^T via 8 WMMAs (HD=128 = 4 K-chunks x 2 key sub-tiles), online
// softmax in C/D layout (row = lane-group of 16 -> xor-shuffle reductions),
// P relayout through per-wave LDS (DS in-order within a wave), PV via 8 WMMAs.
// ---------------------------------------------------------------------------
__global__ __launch_bounds__(128)
void attn_fwd(const unsigned short* __restrict__ Qb,
              const unsigned short* __restrict__ Kb,
              const unsigned short* __restrict__ Vt,
              unsigned short* __restrict__ Y) {
  __shared__ __align__(32) unsigned short pShm[4][16 * 32];
  const int lane = threadIdx.x & 31;
  const int wv   = threadIdx.x >> 5;
  const int w    = blockIdx.x * 4 + wv;   // 0..4095 query blocks
  const int bh   = w >> 7;                // 0..31
  const int qb   = w & 127;
  const int q0   = qb * 16;
  const int half = lane >> 4, ln = lane & 15;
  unsigned short* sp = pShm[wv];

  // Q fragments for this 16-row block: 4 chunks of K=32 over HD=128.
  v16bf qf[4];
#pragma unroll
  for (int c = 0; c < 4; ++c)
    qf[c] = load_a16(Qb + (size_t)(bh * T_ + q0 + ln) * HD_ + c * 32 + half * 16);

  v8f yac[8];
  float mrun[8], lrun[8];
#pragma unroll
  for (int j = 0; j < 8; ++j) { yac[j] = vzero8(); mrun[j] = -1e30f; lrun[j] = 0.0f; }

  const float scale = 0.08838834764831845f;  // 1/sqrt(128)

  for (int kt = 0; kt <= q0 + 15; kt += 32) {
    v8f s0 = vzero8(), s1 = vzero8();
#pragma unroll
    for (int c = 0; c < 4; ++c) {
      v16bf kf0 = load_b16(Kb + (size_t)(bh * T_ + kt + ln) * HD_ + c * 32 + half * 16);
      v16bf kf1 = load_b16(Kb + (size_t)(bh * T_ + kt + 16 + ln) * HD_ + c * 32 + half * 16);
      s0 = wmma_bf16(qf[c], kf0, s0);
      s1 = wmma_bf16(qf[c], kf1, s1);
    }

    float al[8];
#pragma unroll
    for (int j = 0; j < 8; ++j) {
      const int row = q0 + j + 8 * half;       // query index of element j
      const int c0 = kt + ln, c1 = c0 + 16;    // key indices (columns)
      float sv0 = (c0 <= row) ? s0[j] * scale : -1e30f;  // causal mask
      float sv1 = (c1 <= row) ? s1[j] * scale : -1e30f;
      float wmax = fmaxf(sv0, sv1);
#pragma unroll
      for (int d = 1; d < 16; d <<= 1)         // row max within 16-lane half
        wmax = fmaxf(wmax, __shfl_xor(wmax, d, 32));
      const float mnew  = fmaxf(mrun[j], wmax);
      const float alpha = __expf(mrun[j] - mnew);
      const float p0 = __expf(sv0 - mnew);
      const float p1 = __expf(sv1 - mnew);
      float rs = p0 + p1;
#pragma unroll
      for (int d = 1; d < 16; d <<= 1)         // row sum within 16-lane half
        rs += __shfl_xor(rs, d, 32);
      lrun[j] = lrun[j] * alpha + rs;
      mrun[j] = mnew;
      al[j]   = alpha;
      sp[(j + 8 * half) * 32 + ln]      = f2bf(p0);  // stage P in [16][32]
      sp[(j + 8 * half) * 32 + ln + 16] = f2bf(p1);
    }
    asm volatile("" ::: "memory");  // keep DS stores before DS loads (HW keeps per-wave order)
    const v16bf pf = load_a16(sp + ln * 32 + half * 16);

#pragma unroll
    for (int d = 0; d < 8; ++d) {
      const v16bf vf =
          load_b16(Vt + ((size_t)bh * HD_ + d * 16 + ln) * T_ + kt + half * 16);
#pragma unroll
      for (int j = 0; j < 8; ++j) yac[d][j] *= al[j];   // online rescale
      yac[d] = wmma_bf16(pf, vf, yac[d]);
    }
    asm volatile("" ::: "memory");
  }

  // Normalize and store bf16 y as [b, t, h*HD + d] for the output GEMM.
  const int b = bh >> 4, h = bh & 15;
  float inv[8];
#pragma unroll
  for (int j = 0; j < 8; ++j) inv[j] = 1.0f / lrun[j];
#pragma unroll
  for (int d = 0; d < 8; ++d) {
    const int col = h * HD_ + d * 16 + ln;
#pragma unroll
    for (int j = 0; j < 8; ++j) {
      const int row = q0 + j + 8 * half;
      Y[(size_t)(b * T_ + row) * C_ + col] = f2bf(yac[d][j] * inv[j]);
    }
  }
}

// ---------------------------------------------------------------------------
// Launcher. Inputs: x f32 [B,T,C], attn_mask (ignored: causal is analytic),
// W_attn f32 [3C,C], W_proj f32 [C,C]. Output: f32 [B,T,C].
// Workspace usage ~218 MB.
// ---------------------------------------------------------------------------
extern "C" void kernel_launch(void* const* d_in, const int* in_sizes, int n_in,
                              void* d_out, int out_size, void* d_ws, size_t ws_size,
                              hipStream_t stream) {
  (void)in_sizes; (void)n_in; (void)out_size; (void)ws_size;
  const float* x  = (const float*)d_in[0];
  const float* Wa = (const float*)d_in[2];
  const float* Wp = (const float*)d_in[3];
  float* out = (float*)d_out;

  char* ws = (char*)d_ws;
  size_t off = 0;
  auto carve = [&](size_t bytes) -> char* {
    char* p = ws + off;
    off = (off + bytes + 255) & ~(size_t)255;
    return p;
  };

  unsigned short* xh  = (unsigned short*)carve((size_t)BT_ * C_ * 2);
  unsigned short* wa  = (unsigned short*)carve((size_t)NQKV_ * C_ * 2);
  unsigned short* wp  = (unsigned short*)carve((size_t)C_ * C_ * 2);
  float*          qkv = (float*)carve((size_t)BT_ * NQKV_ * 4);
  unsigned short* Qb  = (unsigned short*)carve((size_t)B_ * H_ * T_ * HD_ * 2);
  unsigned short* Kb  = (unsigned short*)carve((size_t)B_ * H_ * T_ * HD_ * 2);
  unsigned short* Vt  = (unsigned short*)carve((size_t)B_ * H_ * T_ * HD_ * 2);
  unsigned short* yA  = (unsigned short*)carve((size_t)BT_ * C_ * 2);

  // 1) precision converts
  cvt_f32_bf16<<<(BT_ * C_ + 255) / 256, 256, 0, stream>>>(x, xh, BT_ * C_);
  cvt_f32_bf16<<<(NQKV_ * C_ + 255) / 256, 256, 0, stream>>>(Wa, wa, NQKV_ * C_);
  cvt_f32_bf16<<<(C_ * C_ + 255) / 256, 256, 0, stream>>>(Wp, wp, C_ * C_);

  // 2) QKV projection: [4096,2048] x [6144,2048]^T -> f32 qkv
  gemm_nt_bf16<<<dim3(NQKV_ / 256, BT_ / 128), 256, 0, stream>>>(
      xh, wa, qkv, BT_, NQKV_, C_);

  // 3) RoPE + pack Q/K [b,h,t,d] and V^T [b,h,d,t] (bf16)
  rope_pack<<<(B_ * T_ * H_ * (HD_ / 2) + 255) / 256, 256, 0, stream>>>(
      qkv, Qb, Kb, Vt);

  // 4) causal flash attention -> bf16 y [b,t,c]
  attn_fwd<<<(B_ * H_ * (T_ / 16)) / 4, 128, 0, stream>>>(Qb, Kb, Vt, yA);

  // 5) output projection: [4096,2048] x [2048,2048]^T -> f32 out
  gemm_nt_bf16<<<dim3(C_ / 256, BT_ / 128), 256, 0, stream>>>(
      yA, wp, out, BT_, C_, C_);
}
